// PoseKeypointGAT_residual_15083925143747
// MI455X (gfx1250) — compile-verified, matrix operand order checked by reading, not run on hardware
//
#include <hip/hip_runtime.h>
#include <hip/hip_fp16.h>

typedef __attribute__((ext_vector_type(16))) _Float16 v16h;
typedef __attribute__((ext_vector_type(8)))  float    v8f;

#define B_TOT   16384
#define NN      17
#define FIN     50
#define HID     128
#define HEADS   4
#define FOUT    50
#define HH      512          // HID*HEADS
#define KFC     850          // NN*FOUT
#define KFC_PAD 864          // padded to 32
#define KT_FC   27           // 864/32
#define NT_FC   54           // 864/16
#define NEG_SLOPE 0.2f
#define LN_EPS  1e-5f

// ---------------------------------------------------------------------------
// WMMA fragment conventions (CDNA5 16-bit operand layout, wave32):
//  A (16Mx32K): lane l (0..15) holds row M=l, K = {kb..kb+7} in elems 0..7 and
//               {kb+16..kb+23} in elems 8..15 with kb = kTile*32; lanes 16..31
//               hold the same rows with kb = kTile*32 + 8.
//  B (32Kx16N): same K mapping, column N = l&15.
//  C/D (v8f):   elem r -> row = r + (l>=16 ? 8 : 0), col = l&15.
// ---------------------------------------------------------------------------

// Gather an A fragment from an f16 LDS activation buffer (row-major, stride lda).
// Rows beyond (rows-1) are clamped: those lanes read valid memory; their
// results are discarded at the C store. Keeps EXEC uniform in the hot loop.
__device__ inline v16h load_a_frag(const _Float16* sA, int lda, int mTile, int kt, int rows) {
    const int lane  = threadIdx.x & 31;
    const int row   = min(mTile * 16 + (lane & 15), rows - 1);
    const int kbase = kt * 32 + ((lane & 16) ? 8 : 0);
    const _Float16* p = sA + row * lda + kbase;
    v16h a;
#pragma unroll
    for (int e = 0; e < 8; ++e) { a[e] = p[e]; a[e + 8] = p[16 + e]; }
    return a;
}

// Generic tiled GEMM: dst[17 x (Nt*16)] = sA[17 x (Kt*32)] * Wp (f32 -> LDS).
// NG N-tiles share one A fragment -> NG independent back-to-back wmmas.
// NOTE: store path is specialized for rows == 17, Mt == 2 (all call sites):
// the m==1 tile holds only row 16 (lanes 0..15, accumulator element 0).
template <int Mt, int Kt, int Nt, int NG>
__device__ inline void gemm_wmma(const _Float16* sA, int lda,
                                 const _Float16* __restrict__ Wp,
                                 float* dst, int ldd, int rows) {
    const int lane = threadIdx.x & 31;
    const int wv   = threadIdx.x >> 5;
    constexpr int NGroups = Nt / NG;
    for (int job = wv; job < Mt * NGroups; job += 8) {
        const int m  = job / NGroups;
        const int n0 = (job % NGroups) * NG;
        v8f acc[NG];
#pragma unroll
        for (int q = 0; q < NG; ++q) acc[q] = (v8f){};
#pragma unroll
        for (int t = 0; t < Kt; ++t) {
            const v16h a = load_a_frag(sA, lda, m, t, rows);
#pragma unroll
            for (int q = 0; q < NG; ++q) {
                const v16h b = *(const v16h*)(Wp + (((size_t)t * Nt + n0 + q) * 32 + lane) * 16);
                acc[q] = __builtin_amdgcn_wmma_f32_16x16x32_f16(false, a, false, b,
                                                               (short)0, acc[q], false, false);
            }
        }
        const int colBase = n0 * 16 + (lane & 15);
        if (m == 0) {
            // rows 0..15 all valid: unguarded stores
            const int rbase = (lane & 16) ? 8 : 0;
#pragma unroll
            for (int q = 0; q < NG; ++q)
#pragma unroll
                for (int r = 0; r < 8; ++r)
                    dst[(rbase + r) * ldd + colBase + q * 16] = acc[q][r];
        } else {
            // m == 1: only row 16 valid -> r == 0 on lanes 0..15
            if ((lane & 16) == 0) {
#pragma unroll
                for (int q = 0; q < NG; ++q)
                    dst[16 * ldd + colBase + q * 16] = acc[q][0];
            }
        }
    }
}

// ---------------------------------------------------------------------------
// Weight pre-pack: f32 [K x N] (or transposed source) -> f16 B-fragment layout
// dst[((t*Nt + n)*32 + lane)*16 + e], zero padded.
// ---------------------------------------------------------------------------
__global__ void pack_w(const float* __restrict__ src, int K, int N, int ld, int trans,
                       int Kt, int Nt, _Float16* __restrict__ dst) {
    const int gid   = blockIdx.x * 256 + threadIdx.x;
    const int total = Kt * Nt * 512;
    if (gid >= total) return;
    const int e    = gid & 15;
    const int l    = (gid >> 4) & 31;
    const int rest = gid >> 9;
    const int n    = rest % Nt;
    const int t    = rest / Nt;
    const int k    = t * 32 + ((l & 16) ? 8 : 0) + (e < 8 ? e : e + 8);
    const int col  = n * 16 + (l & 15);
    float v = 0.f;
    if (k < K && col < N)
        v = trans ? src[(size_t)col * ld + k] : src[(size_t)k * ld + col];
    dst[gid] = (_Float16)v;
}

// ---------------------------------------------------------------------------
// Attention helpers (operate on LDS buffers)
// ---------------------------------------------------------------------------
__device__ inline void attention_heads(float* sg, const _Float16* sA,
                                       const float* __restrict__ as_,
                                       const float* __restrict__ ad_,
                                       const float* __restrict__ bias,
                                       float sed[2][NN][HEADS],
                                       float attw[HEADS][NN][NN],
                                       int add_residual) {
    const int tid = threadIdx.x;
    if (tid < NN * HEADS) {
        const int n = tid >> 2, h = tid & 3;
        float es = 0.f, ed = 0.f;
        const float* hp = &sg[n * HH + h * HID];
        for (int f = 0; f < HID; ++f) {
            const float v = hp[f];
            es += v * as_[h * HID + f];
            ed += v * ad_[h * HID + f];
        }
        sed[0][n][h] = es; sed[1][n][h] = ed;
    }
    __syncthreads();
    if (tid < NN * HEADS) {
        const int i = tid >> 2, h = tid & 3;
        float z[NN];
        float m = -1e30f;
        const float edi = sed[1][i][h];
#pragma unroll
        for (int j = 0; j < NN; ++j) {
            float t = edi + sed[0][j][h];
            t = (t > 0.f) ? t : NEG_SLOPE * t;
            z[j] = t; m = fmaxf(m, t);
        }
        float s = 0.f;
#pragma unroll
        for (int j = 0; j < NN; ++j) { z[j] = __expf(z[j] - m); s += z[j]; }
        const float inv = 1.f / s;
#pragma unroll
        for (int j = 0; j < NN; ++j) attw[h][i][j] = z[j] * inv;
    }
    __syncthreads();
    // per-column weighted sum + bias + relu (+ residual), in place on sg
    for (int c = tid; c < HH; c += 256) {
        const int h = c >> 7;
        float r[NN], o[NN];
#pragma unroll
        for (int j = 0; j < NN; ++j) r[j] = sg[j * HH + c];
        const float bc = bias[c];
#pragma unroll
        for (int i = 0; i < NN; ++i) {
            float acc = bc;
#pragma unroll
            for (int j = 0; j < NN; ++j) acc += attw[h][i][j] * r[j];
            acc = fmaxf(acc, 0.f);
            if (add_residual) acc += (float)sA[i * HH + c];
            o[i] = acc;
        }
#pragma unroll
        for (int i = 0; i < NN; ++i) sg[i * HH + c] = o[i];
    }
    __syncthreads();
}

__device__ inline void layer_norm(const float* sg, _Float16* sA,
                                  const float* __restrict__ g,
                                  const float* __restrict__ be,
                                  float* smu, float* srs) {
    const int tid = threadIdx.x;
    if (tid < NN) {
        float s = 0.f, s2 = 0.f;
        const float* p = &sg[tid * HH];
        for (int c = 0; c < HH; ++c) { const float v = p[c]; s += v; s2 += v * v; }
        const float mu  = s * (1.f / HH);
        const float var = s2 * (1.f / HH) - mu * mu;
        smu[tid] = mu;
        srs[tid] = rsqrtf(var + LN_EPS);
    }
    __syncthreads();
    for (int i = tid; i < NN * HH; i += 256) {
        const int n = i >> 9, c = i & (HH - 1);
        sA[i] = (_Float16)((sg[i] - smu[n]) * srs[n] * g[c] + be[c]);
    }
    __syncthreads();
}

// ---------------------------------------------------------------------------
// Fused 3-layer GAT: one workgroup per batch.
// ---------------------------------------------------------------------------
__global__ __launch_bounds__(256) void gat_fused(
    const float* __restrict__ x,
    const float* __restrict__ a1s, const float* __restrict__ a1d, const float* __restrict__ b1,
    const float* __restrict__ a2s, const float* __restrict__ a2d, const float* __restrict__ b2,
    const float* __restrict__ a3s, const float* __restrict__ a3d, const float* __restrict__ b3,
    const float* __restrict__ g1,  const float* __restrict__ be1,
    const float* __restrict__ g2,  const float* __restrict__ be2,
    const _Float16* __restrict__ W1p, const _Float16* __restrict__ W2p,
    const _Float16* __restrict__ W3p, _Float16* __restrict__ flatA) {

    __shared__ _Float16 sA[NN * HH];          // f16 activations (GEMM A source / residual)
    __shared__ float    sg[NN * HH];          // f32 GEMM / attention buffer
    __shared__ float    sed[2][NN][HEADS];
    __shared__ float    attw[HEADS][NN][NN];
    __shared__ float    smu[NN], srs[NN];

    const int b   = blockIdx.x;
    const int tid = threadIdx.x;

    // ---- load x[b] into sA as f16 [17][64], zero-padded K 50->64 ----
    for (int i = tid; i < NN * 64; i += 256) {
        const int n = i >> 6, k = i & 63;
        sA[i] = (k < FIN) ? (_Float16)x[(size_t)b * NN * FIN + n * FIN + k] : (_Float16)0.f;
    }
    __syncthreads();

    // ---- layer 1: h1 = x @ W1 ----
    gemm_wmma<2, 2, 32, 4>(sA, 64, W1p, sg, HH, NN);
    __syncthreads();
    attention_heads(sg, sA, a1s, a1d, b1, sed, attw, /*residual=*/0);
    layer_norm(sg, sA, g1, be1, smu, srs);

    // ---- layer 2: h2 = h1 @ W2, residual from sA ----
    gemm_wmma<2, 16, 32, 4>(sA, HH, W2p, sg, HH, NN);
    __syncthreads();
    attention_heads(sg, sA, a2s, a2d, b2, sed, attw, /*residual=*/1);
    layer_norm(sg, sA, g2, be2, smu, srs);

    // ---- layer 3: h3 = h2 @ W3 (N padded 50->64), single head, no relu/LN ----
    gemm_wmma<2, 16, 4, 4>(sA, HH, W3p, sg, 64, NN);
    __syncthreads();
    if (tid < NN) {
        float es = 0.f, ed = 0.f;
        const float* hp = &sg[tid * 64];
        for (int f = 0; f < FOUT; ++f) { es += hp[f] * a3s[f]; ed += hp[f] * a3d[f]; }
        sed[0][tid][0] = es; sed[1][tid][0] = ed;
    }
    __syncthreads();
    if (tid < NN) {
        float z[NN];
        float m = -1e30f;
        const float edi = sed[1][tid][0];
#pragma unroll
        for (int j = 0; j < NN; ++j) {
            float t = edi + sed[0][j][0];
            t = (t > 0.f) ? t : NEG_SLOPE * t;
            z[j] = t; m = fmaxf(m, t);
        }
        float s = 0.f;
#pragma unroll
        for (int j = 0; j < NN; ++j) { z[j] = __expf(z[j] - m); s += z[j]; }
        const float inv = 1.f / s;
#pragma unroll
        for (int j = 0; j < NN; ++j) attw[0][tid][j] = z[j] * inv;
    }
    __syncthreads();
    for (int c = tid; c < FOUT; c += 256) {
        float r[NN], o[NN];
#pragma unroll
        for (int j = 0; j < NN; ++j) r[j] = sg[j * 64 + c];
        const float bc = b3[c];
#pragma unroll
        for (int i = 0; i < NN; ++i) {
            float acc = bc;
#pragma unroll
            for (int j = 0; j < NN; ++j) acc += attw[0][i][j] * r[j];
            o[i] = acc;
        }
#pragma unroll
        for (int i = 0; i < NN; ++i) sg[i * 64 + c] = o[i];
    }
    __syncthreads();

    // ---- scatter flat row [850] (pad 864) into WMMA A-fragment layout ----
    {
        const int mb = b >> 4;
        const int rl = b & 15;            // row-within-tile
        for (int k = tid; k < KFC_PAD; k += 256) {
            _Float16 v = (_Float16)0.f;
            if (k < KFC) v = (_Float16)sg[(k / FOUT) * 64 + (k % FOUT)];
            const int t  = k >> 5;
            const int kk = k & 31;
            int lane, e;
            if (kk < 8)       { lane = rl;      e = kk;      }
            else if (kk < 16) { lane = rl + 16; e = kk - 8;  }
            else if (kk < 24) { lane = rl;      e = kk - 8;  }
            else              { lane = rl + 16; e = kk - 16; }
            flatA[(((size_t)mb * KT_FC + t) * 32 + lane) * 16 + e] = v;
        }
    }
}

// ---------------------------------------------------------------------------
// Final FC: out[B,850] = flat[B,850] @ fcw^T + fcb. One WG per 16 batches.
// Pairs of N-tiles share each A fragment; K fully unrolled.
// ---------------------------------------------------------------------------
__global__ __launch_bounds__(256) void fc_kernel(const _Float16* __restrict__ flatA,
                                                 const _Float16* __restrict__ FCWp,
                                                 const float* __restrict__ fcb,
                                                 float* __restrict__ out) {
    const int mb   = blockIdx.x;
    const int lane = threadIdx.x & 31;
    const int wv   = threadIdx.x >> 5;
    for (int jg = wv; jg < NT_FC / 2; jg += 8) {
        const int n0 = jg * 2;
        v8f acc0 = {}, acc1 = {};
#pragma unroll
        for (int t = 0; t < KT_FC; ++t) {
            const v16h a  = *(const v16h*)(flatA + (((size_t)mb * KT_FC + t) * 32 + lane) * 16);
            const v16h b0 = *(const v16h*)(FCWp + (((size_t)t * NT_FC + n0) * 32 + lane) * 16);
            const v16h b1 = *(const v16h*)(FCWp + (((size_t)t * NT_FC + n0 + 1) * 32 + lane) * 16);
            acc0 = __builtin_amdgcn_wmma_f32_16x16x32_f16(false, a, false, b0,
                                                          (short)0, acc0, false, false);
            acc1 = __builtin_amdgcn_wmma_f32_16x16x32_f16(false, a, false, b1,
                                                          (short)0, acc1, false, false);
        }
        const int rbase = (lane & 16) ? 8 : 0;
#pragma unroll
        for (int q = 0; q < 2; ++q) {
            const int col = (n0 + q) * 16 + (lane & 15);
            const v8f& acc = q ? acc1 : acc0;
            if (col < KFC) {
                const float bb = fcb[col];
#pragma unroll
                for (int r = 0; r < 8; ++r) {
                    const int row = mb * 16 + rbase + r;
                    out[(size_t)row * KFC + col] = acc[r] + bb;
                }
            }
        }
    }
}

// ---------------------------------------------------------------------------
extern "C" void kernel_launch(void* const* d_in, const int* in_sizes, int n_in,
                              void* d_out, int out_size, void* d_ws, size_t ws_size,
                              hipStream_t stream) {
    (void)in_sizes; (void)n_in; (void)out_size; (void)ws_size;
    const float* x   = (const float*)d_in[0];
    const float* W1  = (const float*)d_in[1];
    const float* a1s = (const float*)d_in[2];
    const float* a1d = (const float*)d_in[3];
    const float* b1  = (const float*)d_in[4];
    const float* W2  = (const float*)d_in[5];
    const float* a2s = (const float*)d_in[6];
    const float* a2d = (const float*)d_in[7];
    const float* b2  = (const float*)d_in[8];
    const float* W3  = (const float*)d_in[9];
    const float* a3s = (const float*)d_in[10];
    const float* a3d = (const float*)d_in[11];
    const float* b3  = (const float*)d_in[12];
    const float* g1  = (const float*)d_in[13];
    const float* be1 = (const float*)d_in[14];
    const float* g2  = (const float*)d_in[15];
    const float* be2 = (const float*)d_in[16];
    const float* fcw = (const float*)d_in[17];
    const float* fcb = (const float*)d_in[18];
    float* out = (float*)d_out;

    // workspace layout (f16 element counts)
    _Float16* W1p   = (_Float16*)d_ws;                 //  2*32*512 = 32768
    _Float16* W2p   = W1p + 2 * 32 * 512;              // 16*32*512 = 262144
    _Float16* W3p   = W2p + 16 * 32 * 512;             // 16* 4*512 = 32768
    _Float16* FCWp  = W3p + 16 * 4 * 512;              // 27*54*512 = 746496
    _Float16* flatA = FCWp + (size_t)KT_FC * NT_FC * 512; // 1024*27*512 halfs

    pack_w<<<(2 * 32 * 512 + 255) / 256, 256, 0, stream>>>(W1, FIN, HH, HH, 0, 2, 32, W1p);
    pack_w<<<(16 * 32 * 512 + 255) / 256, 256, 0, stream>>>(W2, HH, HH, HH, 0, 16, 32, W2p);
    pack_w<<<(16 * 4 * 512 + 255) / 256, 256, 0, stream>>>(W3, HH, FOUT, FOUT, 0, 16, 4, W3p);
    pack_w<<<(KT_FC * NT_FC * 512 + 255) / 256, 256, 0, stream>>>(fcw, KFC, KFC, KFC, 1, KT_FC, NT_FC, FCWp);

    gat_fused<<<B_TOT, 256, 0, stream>>>(x, a1s, a1d, b1, a2s, a2d, b2, a3s, a3d, b3,
                                         g1, be1, g2, be2, W1p, W2p, W3p, flatA);

    fc_kernel<<<B_TOT / 16, 256, 0, stream>>>(flatA, FCWp, fcb, out);
}
